// TransformerBlock_41334765257047
// MI455X (gfx1250) — matrix operand order repaired and checked by reading, not compiled, not run on hardware
//
#include <hip/hip_runtime.h>
#include <hip/hip_bf16.h>
#include <math.h>
#include <stdint.h>

// ---------------- shapes (fixed by reference) ----------------
#define BATCH 2
#define SEQ   2048
#define DMODEL 512
#define NHEAD 8
#define HDIM  64
#define DFF   2048
#define MTOK  (BATCH*SEQ)          // 4096 token rows

typedef __attribute__((ext_vector_type(16))) __bf16 bf16x16;
typedef __attribute__((ext_vector_type(8)))  float  f32x8;
typedef __attribute__((ext_vector_type(4)))  unsigned int u32x4;
typedef __attribute__((ext_vector_type(8)))  int i32x8;
typedef __attribute__((ext_vector_type(4)))  int i32x4;

union AF { bf16x16 v; uint4 q[2]; };

// ---------------- helpers ----------------
__device__ inline float rmax16(float x) {
  x = fmaxf(x, __shfl_xor(x, 1, 32));
  x = fmaxf(x, __shfl_xor(x, 2, 32));
  x = fmaxf(x, __shfl_xor(x, 4, 32));
  x = fmaxf(x, __shfl_xor(x, 8, 32));
  return x;
}
__device__ inline float rsum16(float x) {
  x += __shfl_xor(x, 1, 32);
  x += __shfl_xor(x, 2, 32);
  x += __shfl_xor(x, 4, 32);
  x += __shfl_xor(x, 8, 32);
  return x;
}

// ---- Tensor Data Mover: DMA a (rows x 32 bf16) tile, row stride = srcStride
// elements, into LDS at lds_addr with +16B padding every 64B (row stride 40
// elements in LDS). Descriptor per CDNA5 ISA 8.3/8.4. Wave-level op.
__device__ inline void tdm_load_tile(const __bf16* gptr, unsigned lds_addr,
                                     int rows, int srcStride) {
  unsigned long long ga = (unsigned long long)(uintptr_t)gptr;
  u32x4 g0;
  g0.x = 1u;                                               // count=1, user mode
  g0.y = lds_addr;                                         // LDS byte address
  g0.z = (unsigned)(ga & 0xFFFFFFFFu);                     // global_addr[31:0]
  g0.w = (unsigned)((ga >> 32) & 0x01FFFFFFu) | 0x80000000u; // [56:32] | type=2
  i32x8 g1;
  // data_size=2B(code 1), pad_enable, pad_interval=16DW(code 3), pad_amount=4DW(code 3)
  g1[0] = (1 << 16) | (1 << 20) | (3 << 22) | (3 << 25);
  g1[1] = (srcStride & 0xFFFF) << 16;                      // tensor_dim0[15:0] @ bits63:48
  g1[2] = ((srcStride >> 16) & 0xFFFF) | ((rows & 0xFFFF) << 16); // dim0 hi | tensor_dim1 lo
  g1[3] = ((rows >> 16) & 0xFFFF) | (32 << 16);            // tensor_dim1 hi | tile_dim0=32
  g1[4] = rows & 0xFFFF;                                   // tile_dim1 (tile_dim2=0)
  g1[5] = srcStride;                                       // tensor_dim0_stride[31:0]
  g1[6] = 0;                                               // stride hi | dim1_stride lo
  g1[7] = 0;
  i32x4 z4 = {0, 0, 0, 0};
#if __has_include(<hip/amd_detail/amd_gfx1250_TDM.h>)
  i32x8 z8 = {0, 0, 0, 0, 0, 0, 0, 0};
  __builtin_amdgcn_tensor_load_to_lds(g0, g1, z4, z4, z8, 0);
#else
  __builtin_amdgcn_tensor_load_to_lds(g0, g1, z4, z4, 0);
#endif
}

// ---------------- fp32 -> bf16 convert (weights) ----------------
__global__ __launch_bounds__(256) void f2bf_kernel(const float* __restrict__ src,
                                                   __bf16* __restrict__ dst, int n) {
  int i = blockIdx.x * blockDim.x + threadIdx.x;
  if (i < n) dst[i] = (__bf16)src[i];
}

// ---------------- RMSNorm (one block per token row of 512) ----------------
__global__ __launch_bounds__(128) void rmsnorm_kernel(const float* __restrict__ x,
                                                      const float* __restrict__ scale,
                                                      __bf16* __restrict__ out) {
  const int row = blockIdx.x;
  const int tid = threadIdx.x;
  const float4 xv = ((const float4*)(x + (size_t)row * DMODEL))[tid];
  float ss = xv.x*xv.x + xv.y*xv.y + xv.z*xv.z + xv.w*xv.w;
  #pragma unroll
  for (int m = 1; m < 32; m <<= 1) ss += __shfl_xor(ss, m, 32);
  __shared__ float red[4];
  if ((tid & 31) == 0) red[tid >> 5] = ss;
  __syncthreads();
  ss = red[0] + red[1] + red[2] + red[3];
  const float r = rsqrtf(ss * (1.0f / DMODEL) + 1e-8f);
  const float4 sv = ((const float4*)scale)[tid];
  __bf16* o = out + (size_t)row * DMODEL + tid * 4;
  o[0] = (__bf16)(sv.x * xv.x * r);
  o[1] = (__bf16)(sv.y * xv.y * r);
  o[2] = (__bf16)(sv.z * xv.z * r);
  o[3] = (__bf16)(sv.w * xv.w * r);
}

// ---------------- generic GEMM: C = A(MxK) * W(NxK)^T + bias, WMMA bf16 ----------------
// Block: 256 threads = 8 waves; tile M=128 (16/wave), N=64.
// B tile (64x32) double-buffered in LDS via TDM tensor_load_to_lds (wave 0 issues,
// hardware pads rows to 40-element stride -> conflict-free ds_load_b128 frags).
// MODE 0: QKV scatter (q scaled, v transposed)  MODE 1: out-proj + residual (fp32)
// MODE 2: GELU -> bf16                          MODE 3: FFN2 + residual (fp32)
template <int MODE>
__global__ __launch_bounds__(256) void gemm_bf16_kernel(
    const __bf16* __restrict__ A, const __bf16* __restrict__ W,
    const float* __restrict__ bias, int N, int K,
    const float* __restrict__ resid, float* __restrict__ outF,
    __bf16* __restrict__ outH,
    __bf16* __restrict__ qO, __bf16* __restrict__ kO, __bf16* __restrict__ vO) {
  const int lane = threadIdx.x & 31;
  const int wave = threadIdx.x >> 5;
  const int half = lane >> 4;
  const int ln16 = lane & 15;
  const int mBase = blockIdx.y * 128 + wave * 16;
  const int nBase = blockIdx.x * 64;

  __shared__ __bf16 bsh[2][64 * 40];  // two staged tiles, stride 40 (80B) rows

  f32x8 acc[4] = {{}, {}, {}, {}};
  const __bf16* arow  = A + (size_t)(mBase + ln16) * K;
  const __bf16* wtile = W + (size_t)nBase * K;
  const unsigned lds0 = (unsigned)(uintptr_t)&bsh[0][0];
  const unsigned ldsStep = 64u * 40u * 2u;

  if (wave == 0) tdm_load_tile(wtile, lds0, 64, K);  // prologue tile 0
  int buf = 0;

  for (int kk = 0; kk < K; kk += 32) {
    if (wave == 0) __builtin_amdgcn_s_wait_tensorcnt(0);
    __syncthreads();                     // current tile visible to all waves
    if (wave == 0 && kk + 32 < K)        // kick next tile into other buffer
      tdm_load_tile(wtile + kk + 32, lds0 + (buf ^ 1) * ldsStep, 64, K);

    AF a;
    const __bf16* ap = arow + kk + half * 8;
    a.q[0] = *(const uint4*)(ap);
    a.q[1] = *(const uint4*)(ap + 16);
    if (kk + 32 < K)                     // CDNA5 global_prefetch_b8
      __builtin_prefetch(arow + kk + 32, 0, 1);

    #pragma unroll
    for (int t = 0; t < 4; ++t) {
      AF b;
      const __bf16* wp = &bsh[buf][(t * 16 + ln16) * 40 + half * 8];
      b.q[0] = *(const uint4*)(wp);
      b.q[1] = *(const uint4*)(wp + 16);
      acc[t] = __builtin_amdgcn_wmma_f32_16x16x32_bf16(false, a.v, false, b.v,
                                                       (short)0, acc[t], false, false);
    }
    buf ^= 1;
  }

  #pragma unroll
  for (int t = 0; t < 4; ++t) {
    #pragma unroll
    for (int v = 0; v < 8; ++v) {
      const int m = mBase + v + half * 8;
      const int n = nBase + t * 16 + ln16;
      float val = acc[t][v] + bias[n];
      if (MODE == 0) {
        const int bb = m >> 11, s = m & (SEQ - 1);
        if (n < DMODEL) {
          const int hh = n >> 6, d = n & 63;
          qO[(((size_t)bb * NHEAD + hh) * SEQ + s) * HDIM + d] = (__bf16)(val * 0.125f);
        } else if (n < 2 * DMODEL) {
          const int f = n - DMODEL, hh = f >> 6, d = f & 63;
          kO[(((size_t)bb * NHEAD + hh) * SEQ + s) * HDIM + d] = (__bf16)val;
        } else {
          const int f = n - 2 * DMODEL, hh = f >> 6, d = f & 63;
          vO[(((size_t)bb * NHEAD + hh) * HDIM + d) * SEQ + s] = (__bf16)val;  // V^T
        }
      } else if (MODE == 1 || MODE == 3) {
        outF[(size_t)m * N + n] = resid[(size_t)m * N + n] + val;
      } else {  // MODE 2 : exact GELU
        float g = 0.5f * val * (1.0f + erff(val * 0.70710678118f));
        outH[(size_t)m * N + n] = (__bf16)g;
      }
    }
  }
}

// ---------------- flash attention with ALiBi: one wave per 16-query tile ----------------
// Key tile = 64 -> 4 score tiles per iteration.
__global__ __launch_bounds__(32) void attn_kernel(const __bf16* __restrict__ q,
                                                  const __bf16* __restrict__ k,
                                                  const __bf16* __restrict__ vt,
                                                  __bf16* __restrict__ ctx) {
  const int lane = threadIdx.x & 31;
  const int half = lane >> 4;
  const int ln16 = lane & 15;
  const int bh = blockIdx.x;         // b*NHEAD + h
  const int h  = bh & (NHEAD - 1);
  const int b  = bh >> 3;
  const int q0 = blockIdx.y * 16;
  const float slope = 1.0f / (float)(h + 1);

  __shared__ __bf16 pbuf[16 * 72];   // 16 rows x 64 cols, stride 72 (144B, 16B aligned)

  // preload Q A-fragments (16x64 = two K=32 chunks), 1/sqrt(64) already folded in
  AF qa[2];
  #pragma unroll
  for (int c = 0; c < 2; ++c) {
    const __bf16* base = q + ((size_t)bh * SEQ + (q0 + ln16)) * HDIM + c * 32 + half * 8;
    qa[c].q[0] = *(const uint4*)(base);
    qa[c].q[1] = *(const uint4*)(base + 16);
  }

  f32x8 o[4] = {{}, {}, {}, {}};
  float mrow[8], lrow[8];
  #pragma unroll
  for (int v = 0; v < 8; ++v) { mrow[v] = -INFINITY; lrow[v] = 0.0f; }

  for (int j = 0; j < SEQ; j += 64) {
    // --- scores: (16 q) x (64 keys), K = 64 head dims ---
    f32x8 s[4] = {{}, {}, {}, {}};
    #pragma unroll
    for (int c = 0; c < 2; ++c) {
      #pragma unroll
      for (int t = 0; t < 4; ++t) {
        AF bk;
        const __bf16* p =
            k + ((size_t)bh * SEQ + (j + t * 16 + ln16)) * HDIM + c * 32 + half * 8;
        bk.q[0] = *(const uint4*)(p);
        bk.q[1] = *(const uint4*)(p + 16);
        s[t] = __builtin_amdgcn_wmma_f32_16x16x32_bf16(false, qa[c].v, false, bk.v,
                                                       (short)0, s[t], false, false);
      }
    }

    __syncthreads();  // protect pbuf WAR vs previous iteration's reads

    // --- ALiBi bias + online softmax update ---
    float fac[8];
    #pragma unroll
    for (int v = 0; v < 8; ++v) {
      const int row = q0 + v + half * 8;
      float a0 = s[0][v] - slope * fabsf((float)(row - (j + ln16)));
      float a1 = s[1][v] - slope * fabsf((float)(row - (j + 16 + ln16)));
      float a2 = s[2][v] - slope * fabsf((float)(row - (j + 32 + ln16)));
      float a3 = s[3][v] - slope * fabsf((float)(row - (j + 48 + ln16)));
      float tmax = fmaxf(fmaxf(a0, a1), fmaxf(a2, a3));
      float nm = fmaxf(mrow[v], rmax16(tmax));
      float f  = __expf(mrow[v] - nm);
      float p0 = __expf(a0 - nm);
      float p1 = __expf(a1 - nm);
      float p2 = __expf(a2 - nm);
      float p3 = __expf(a3 - nm);
      lrow[v] = lrow[v] * f + rsum16(p0 + p1 + p2 + p3);
      mrow[v] = nm;
      fac[v] = f;
      __bf16* pr = pbuf + (v + half * 8) * 72;
      pr[ln16]      = (__bf16)p0;
      pr[16 + ln16] = (__bf16)p1;
      pr[32 + ln16] = (__bf16)p2;
      pr[48 + ln16] = (__bf16)p3;
    }
    #pragma unroll
    for (int v = 0; v < 8; ++v) {
      o[0][v] *= fac[v]; o[1][v] *= fac[v]; o[2][v] *= fac[v]; o[3][v] *= fac[v];
    }
    __syncthreads();  // pbuf RAW

    // --- O += P(16x64) * V(64x64), V stored transposed (d, s) ---
    AF pa[2];
    #pragma unroll
    for (int c = 0; c < 2; ++c) {
      const __bf16* pp = pbuf + ln16 * 72 + c * 32 + half * 8;
      pa[c].q[0] = *(const uint4*)(pp);
      pa[c].q[1] = *(const uint4*)(pp + 16);
    }
    #pragma unroll
    for (int c = 0; c < 2; ++c) {
      #pragma unroll
      for (int t = 0; t < 4; ++t) {
        AF bv;
        const __bf16* vb =
            vt + ((size_t)bh * HDIM + (t * 16 + ln16)) * SEQ + j + c * 32 + half * 8;
        bv.q[0] = *(const uint4*)(vb);
        bv.q[1] = *(const uint4*)(vb + 16);
        o[t] = __builtin_amdgcn_wmma_f32_16x16x32_bf16(false, pa[c].v, false, bv.v,
                                                       (short)0, o[t], false, false);
      }
    }
  }

  // --- normalize + write ctx as (b, s, h*64+d) bf16 ---
  #pragma unroll
  for (int v = 0; v < 8; ++v) {
    const int row = q0 + v + half * 8;
    const float inv = 1.0f / lrow[v];
    __bf16* dst = ctx + ((size_t)b * SEQ + row) * DMODEL + h * HDIM;
    #pragma unroll
    for (int t = 0; t < 4; ++t) dst[t * 16 + ln16] = (__bf16)(o[t][v] * inv);
  }
}

// ---------------- host side ----------------
extern "C" void kernel_launch(void* const* d_in, const int* in_sizes, int n_in,
                              void* d_out, int out_size, void* d_ws, size_t ws_size,
                              hipStream_t stream) {
  (void)in_sizes; (void)n_in; (void)out_size; (void)ws_size;
  const float* x     = (const float*)d_in[0];
  const float* n1s   = (const float*)d_in[1];
  const float* n2s   = (const float*)d_in[2];
  const float* wqkv  = (const float*)d_in[3];
  const float* bqkv  = (const float*)d_in[4];
  const float* wo    = (const float*)d_in[5];
  const float* bo    = (const float*)d_in[6];
  const float* w1    = (const float*)d_in[7];
  const float* b1    = (const float*)d_in[8];
  const float* w2    = (const float*)d_in[9];
  const float* b2    = (const float*)d_in[10];
  float* out = (float*)d_out;

  char* ws = (char*)d_ws;
  size_t off = 0;
  auto take = [&](size_t bytes) -> char* {
    char* p = ws + off;
    off = (off + bytes + 255) & ~(size_t)255;
    return p;
  };
  __bf16* wqkv_h = (__bf16*)take((size_t)3 * DMODEL * DMODEL * 2);
  __bf16* wo_h   = (__bf16*)take((size_t)DMODEL * DMODEL * 2);
  __bf16* w1_h   = (__bf16*)take((size_t)DFF * DMODEL * 2);
  __bf16* w2_h   = (__bf16*)take((size_t)DMODEL * DFF * 2);
  __bf16* xn     = (__bf16*)take((size_t)MTOK * DMODEL * 2);
  __bf16* qb     = (__bf16*)take((size_t)MTOK * DMODEL * 2);
  __bf16* kb     = (__bf16*)take((size_t)MTOK * DMODEL * 2);
  __bf16* vtb    = (__bf16*)take((size_t)MTOK * DMODEL * 2);
  __bf16* ctxb   = (__bf16*)take((size_t)MTOK * DMODEL * 2);
  float*  xres   = (float*) take((size_t)MTOK * DMODEL * 4);
  __bf16* xn2    = (__bf16*)take((size_t)MTOK * DMODEL * 2);
  __bf16* hb     = (__bf16*)take((size_t)MTOK * DFF * 2);

  // 1) weight conversion fp32 -> bf16
  int nw;
  nw = 3 * DMODEL * DMODEL; f2bf_kernel<<<(nw + 255) / 256, 256, 0, stream>>>(wqkv, wqkv_h, nw);
  nw = DMODEL * DMODEL;     f2bf_kernel<<<(nw + 255) / 256, 256, 0, stream>>>(wo, wo_h, nw);
  nw = DFF * DMODEL;        f2bf_kernel<<<(nw + 255) / 256, 256, 0, stream>>>(w1, w1_h, nw);
  nw = DMODEL * DFF;        f2bf_kernel<<<(nw + 255) / 256, 256, 0, stream>>>(w2, w2_h, nw);

  // 2) RMSNorm1
  rmsnorm_kernel<<<MTOK, 128, 0, stream>>>(x, n1s, xn);

  // 3) QKV projection + scatter (M=4096, N=1536, K=512)
  gemm_bf16_kernel<0><<<dim3(1536 / 64, MTOK / 128), 256, 0, stream>>>(
      xn, wqkv_h, bqkv, 1536, DMODEL, nullptr, nullptr, nullptr, qb, kb, vtb);

  // 4) attention (one wave per 16 queries per head)
  attn_kernel<<<dim3(BATCH * NHEAD, SEQ / 16), 32, 0, stream>>>(qb, kb, vtb, ctxb);

  // 5) out-proj + residual (M=4096, N=512, K=512)
  gemm_bf16_kernel<1><<<dim3(DMODEL / 64, MTOK / 128), 256, 0, stream>>>(
      ctxb, wo_h, bo, DMODEL, DMODEL, x, xres, nullptr, nullptr, nullptr, nullptr);

  // 6) RMSNorm2
  rmsnorm_kernel<<<MTOK, 128, 0, stream>>>(xres, n2s, xn2);

  // 7) FFN1 + GELU (M=4096, N=2048, K=512)
  gemm_bf16_kernel<2><<<dim3(DFF / 64, MTOK / 128), 256, 0, stream>>>(
      xn2, w1_h, b1, DFF, DMODEL, nullptr, nullptr, hb, nullptr, nullptr, nullptr);

  // 8) FFN2 + residual -> output (M=4096, N=512, K=2048)
  gemm_bf16_kernel<3><<<dim3(DMODEL / 64, MTOK / 128), 256, 0, stream>>>(
      hb, w2_h, b2, DMODEL, DFF, xres, out, nullptr, nullptr, nullptr, nullptr);
}
